// HeteroGCN_6622839570445
// MI455X (gfx1250) — matrix operand order repaired and checked by reading, not compiled, not run on hardware
//
#include <hip/hip_runtime.h>
#include <stdint.h>

// ---------------------------------------------------------------------------
// HeteroGCN for MI455X (gfx1250, wave32).
// Dense transforms: v_wmma_f32_16x16x32_bf16 (bf16 in, f32 accumulate),
//   one wave computes a 16x64 output strip (4 N-tiles) reusing the A operand.
// Edge aggregation: wave-per-edge vectorized gather + f32 atomic scatter-add
//   (feature matrices are L2-resident on the 192MB L2).
// ---------------------------------------------------------------------------

typedef __bf16 bf16;
typedef __attribute__((ext_vector_type(16))) __bf16 v16bf;
typedef __attribute__((ext_vector_type(8)))  __bf16 v8bf;
typedef __attribute__((ext_vector_type(8)))  float  v8f;

#define NP_ 100000
#define NL_ 64

// ---------------------------- elementwise helpers ---------------------------

__global__ void k_zero(float* p, int n) {
    int i = blockIdx.x * blockDim.x + threadIdx.x;
    if (i < n) p[i] = 0.0f;
}

__global__ void k_deg(const int* __restrict__ dst, float* __restrict__ deg, int E) {
    int i = blockIdx.x * blockDim.x + threadIdx.x;
    if (i < E) atomicAdd(&deg[dst[i]], 1.0f);
}

__global__ void k_rsqrt1(float* d, int n) {  // d[i] = rsqrt(deg[i] + 1)  (self loop)
    int i = blockIdx.x * blockDim.x + threadIdx.x;
    if (i < n) d[i] = rsqrtf(d[i] + 1.0f);
}

__global__ void k_f2b(const float* __restrict__ x, bf16* __restrict__ y, int n) {
    int i = blockIdx.x * blockDim.x + threadIdx.x;
    if (i < n) y[i] = (bf16)x[i];
}

// Wt[n*K + k] = bf16(W[k*N + n])  (row-major [K,N] -> transposed bf16 [N,K])
__global__ void k_wt(const float* __restrict__ W, bf16* __restrict__ Wt, int K, int N) {
    int i = blockIdx.x * blockDim.x + threadIdx.x;
    if (i < K * N) {
        int k = i / N, n = i % N;
        Wt[(size_t)n * K + k] = (bf16)W[i];
    }
}

// agg[i] = dis[r]*agg[i] + dis[r]^2*y[i] + bias[c]   (GCNConv epilogue, in place)
__global__ void k_gcn_combine(float* __restrict__ agg, const float* __restrict__ y,
                              const float* __restrict__ dis, const float* __restrict__ bias,
                              int n, int F) {
    int i = blockIdx.x * blockDim.x + threadIdx.x;
    if (i >= n * F) return;
    int r = i / F, c = i % F;
    float dv = dis[r];
    agg[i] = dv * agg[i] + dv * dv * y[i] + bias[c];
}

// v = relu(0.5*(a+b)); of = v; ob = bf16(v)
__global__ void k_mean_relu(const float* __restrict__ a, const float* __restrict__ b,
                            float* __restrict__ of, bf16* __restrict__ ob, int n) {
    int i = blockIdx.x * blockDim.x + threadIdx.x;
    if (i >= n) return;
    float v = 0.5f * (a[i] + b[i]);
    v = v > 0.0f ? v : 0.0f;
    of[i] = v;
    ob[i] = (bf16)v;
}

// in-place relu + bf16 copy
__global__ void k_relu_conv(float* __restrict__ a, bf16* __restrict__ ob, int n) {
    int i = blockIdx.x * blockDim.x + threadIdx.x;
    if (i >= n) return;
    float v = a[i];
    v = v > 0.0f ? v : 0.0f;
    a[i] = v;
    ob[i] = (bf16)v;
}

__global__ void k_mean(const float* __restrict__ a, const float* __restrict__ b,
                       float* __restrict__ o, int n) {
    int i = blockIdx.x * blockDim.x + threadIdx.x;
    if (i < n) o[i] = 0.5f * (a[i] + b[i]);
}

// --------------------------- edge scatter (wave/edge) -----------------------
// agg[dst[e], :] += x[src[e], :] * (scale ? scale[src[e]] : 1)
// F == 128: one float4 load + 4 atomic adds per lane.
__global__ void k_scatter128(const float* __restrict__ x, const int* __restrict__ src,
                             const int* __restrict__ dst, const float* __restrict__ scale,
                             float* __restrict__ agg, int E) {
    int gid  = blockIdx.x * blockDim.x + threadIdx.x;
    int e    = gid >> 5;
    int lane = gid & 31;
    if (e >= E) return;
    int s = src[e], d = dst[e];
    float sc = scale ? scale[s] : 1.0f;
    const float4 v = *(const float4*)(x + (size_t)s * 128 + lane * 4);
    float* ar = agg + (size_t)d * 128 + lane * 4;
    atomicAdd(ar + 0, v.x * sc);
    atomicAdd(ar + 1, v.y * sc);
    atomicAdd(ar + 2, v.z * sc);
    atomicAdd(ar + 3, v.w * sc);
}

// F == 64: one float2 load + 2 atomic adds per lane.
__global__ void k_scatter64(const float* __restrict__ x, const int* __restrict__ src,
                            const int* __restrict__ dst, const float* __restrict__ scale,
                            float* __restrict__ agg, int E) {
    int gid  = blockIdx.x * blockDim.x + threadIdx.x;
    int e    = gid >> 5;
    int lane = gid & 31;
    if (e >= E) return;
    int s = src[e], d = dst[e];
    float sc = scale ? scale[s] : 1.0f;
    const float2 v = *(const float2*)(x + (size_t)s * 64 + lane * 2);
    float* ar = agg + (size_t)d * 64 + lane * 2;
    atomicAdd(ar + 0, v.x * sc);
    atomicAdd(ar + 1, v.y * sc);
}

// ------------------------------- WMMA GEMM ----------------------------------
// C[M,N] (f32) = A[M,K](bf16, row-major) @ Bt[N,K](bf16, row-major = B^T)
//               (+ bias[N]) (+= if accum)
// One wave computes a 16x64 strip: the A operand is loaded once per K-step and
// reused across 4 consecutive 16x16 N-tiles (4 WMMAs per A fetch).
// Requires M%16==0, N%64==0, K%32==0 (all shapes here satisfy this).
__global__ void k_gemm_bf16(const bf16* __restrict__ A, const bf16* __restrict__ Bt,
                            float* __restrict__ C, const float* __restrict__ bias,
                            int M, int N, int K, int accum) {
    int warp = threadIdx.x >> 5;
    int lane = threadIdx.x & 31;
    int nStrips = N >> 6;                       // strips of 64 columns
    int strips  = (M >> 4) * nStrips;
    int strip   = blockIdx.x * (blockDim.x >> 5) + warp;
    if (strip >= strips) return;                // whole wave exits together

    int mT = strip / nStrips;
    int nS = strip % nStrips;
    int l15  = lane & 15;
    int half = lane >> 4;

    // A operand: lane l15 holds row (mT*16+l15); lane-half selects K sub-block.
    //   elems e -> K = k0 + half*16 + e
    const bf16* Ap = A + (size_t)((mT << 4) + l15) * K + half * 16;
    // B operand: lane l15 holds column; elems e -> K = k0 + half*16 + e
    const bf16* Bp0 = Bt + (size_t)((nS << 6) + l15) * K + half * 16;

    v8f acc0 = {}, acc1 = {}, acc2 = {}, acc3 = {};
    for (int k0 = 0; k0 < K; k0 += 32) {
        v8bf alo = *(const v8bf*)(Ap + k0);
        v8bf ahi = *(const v8bf*)(Ap + k0 + 8);
        v16bf a = __builtin_shufflevector(alo, ahi,
                    0, 1, 2, 3, 4, 5, 6, 7, 8, 9, 10, 11, 12, 13, 14, 15);
        v16bf b0 = *(const v16bf*)(Bp0 + k0);
        v16bf b1 = *(const v16bf*)(Bp0 + (size_t)16 * K + k0);
        v16bf b2 = *(const v16bf*)(Bp0 + (size_t)32 * K + k0);
        v16bf b3 = *(const v16bf*)(Bp0 + (size_t)48 * K + k0);
        acc0 = __builtin_amdgcn_wmma_f32_16x16x32_bf16(false, a, false, b0, (short)0, acc0, false, false);
        acc1 = __builtin_amdgcn_wmma_f32_16x16x32_bf16(false, a, false, b1, (short)0, acc1, false, false);
        acc2 = __builtin_amdgcn_wmma_f32_16x16x32_bf16(false, a, false, b2, (short)0, acc2, false, false);
        acc3 = __builtin_amdgcn_wmma_f32_16x16x32_bf16(false, a, false, b3, (short)0, acc3, false, false);
    }

    int rbase = (mT << 4) + (half << 3);        // acc elem r -> row rbase + r
    v8f* accs[4] = {&acc0, &acc1, &acc2, &acc3};
#pragma unroll
    for (int nt = 0; nt < 4; ++nt) {
        int col = (nS << 6) + (nt << 4) + l15;
        float bi = bias ? bias[col] : 0.0f;
        v8f acc = *accs[nt];
#pragma unroll
        for (int r = 0; r < 8; ++r) {
            size_t idx = (size_t)(rbase + r) * N + col;
            float v = acc[r] + bi;
            if (accum) C[idx] += v; else C[idx] = v;
        }
    }
}

// ------------------------------- host driver --------------------------------

static inline int cdiv(long a, long b) { return (int)((a + b - 1) / b); }

static void gemm(const bf16* A, const bf16* Bt, float* C, const float* bias,
                 int M, int N, int K, int accum, hipStream_t s) {
    int strips = (M >> 4) * (N >> 6);
    k_gemm_bf16<<<cdiv(strips, 8), 256, 0, s>>>(A, Bt, C, bias, M, N, K, accum);
}

extern "C" void kernel_launch(void* const* d_in, const int* in_sizes, int n_in,
                              void* d_out, int out_size, void* d_ws, size_t ws_size,
                              hipStream_t stream) {
    (void)n_in; (void)out_size; (void)ws_size;
    const float* x_paper = (const float*)d_in[0];
    const float* x_label = (const float*)d_in[1];
    const float* W1g  = (const float*)d_in[2];  const float* b1g = (const float*)d_in[3];
    const float* W1ri = (const float*)d_in[4];  const float* W1oi = (const float*)d_in[5];
    const float* b1i  = (const float*)d_in[6];
    const float* W1rr = (const float*)d_in[7];  const float* W1or = (const float*)d_in[8];
    const float* b1r  = (const float*)d_in[9];
    const float* W2g  = (const float*)d_in[10]; const float* b2g = (const float*)d_in[11];
    const float* W2ri = (const float*)d_in[12]; const float* W2oi = (const float*)d_in[13];
    const float* b2i  = (const float*)d_in[14];
    const float* W2rr = (const float*)d_in[15]; const float* W2or = (const float*)d_in[16];
    const float* b2r  = (const float*)d_in[17];
    const int* cites_src = (const int*)d_in[18];
    const int* cites_dst = (const int*)d_in[19];
    const int* is_src    = (const int*)d_in[20];
    const int* is_dst    = (const int*)d_in[21];
    const int* rev_src   = (const int*)d_in[22];
    const int* rev_dst   = (const int*)d_in[23];
    const int EC = in_sizes[18], EI = in_sizes[20], ER = in_sizes[22];

    // -------- workspace carve-out (256B aligned) --------
    char* base = (char*)d_ws;
    size_t off = 0;
    auto carve = [&](size_t bytes) -> char* {
        char* p = base + off;
        off += (bytes + 255) & ~(size_t)255;
        return p;
    };
    float* dis   = (float*)carve((size_t)NP_ * 4);            // deg -> rsqrt(deg+1)
    bf16*  xpb   = (bf16*) carve((size_t)NP_ * 128 * 2);
    bf16*  xlb   = (bf16*) carve((size_t)NL_ * 64 * 2);
    bf16* w1g_t  = (bf16*) carve(128 * 128 * 2);
    bf16* w1ri_t = (bf16*) carve(128 * 128 * 2);
    bf16* w1oi_t = (bf16*) carve(64  * 128 * 2);
    bf16* w1rr_t = (bf16*) carve(64  * 128 * 2);
    bf16* w1or_t = (bf16*) carve(128 * 128 * 2);
    bf16* w2g_t  = (bf16*) carve(128 * 64 * 2);
    bf16* w2ri_t = (bf16*) carve(128 * 64 * 2);
    bf16* w2oi_t = (bf16*) carve(128 * 64 * 2);
    bf16* w2rr_t = (bf16*) carve(128 * 64 * 2);
    bf16* w2or_t = (bf16*) carve(128 * 64 * 2);
    float* yg    = (float*)carve((size_t)NP_ * 128 * 4);
    float* aggP  = (float*)carve((size_t)NP_ * 128 * 4);
    float* prev  = (float*)carve((size_t)NP_ * 128 * 4);
    float* aggRv = (float*)carve((size_t)NP_ * 128 * 4);
    bf16*  aggRvB= (bf16*) carve((size_t)NP_ * 128 * 2);
    float* aggIs = (float*)carve((size_t)NL_ * 128 * 4);
    bf16*  aggIsB= (bf16*) carve((size_t)NL_ * 128 * 2);
    float* xp1   = (float*)carve((size_t)NP_ * 128 * 4);
    bf16*  xp1b  = (bf16*) carve((size_t)NP_ * 128 * 2);
    float* xl1   = (float*)carve((size_t)NL_ * 128 * 4);
    bf16*  xl1b  = (bf16*) carve((size_t)NL_ * 128 * 2);

    float* out_p = (float*)d_out;                     // [NP, 64]
    float* out_l = (float*)d_out + (size_t)NP_ * 64;  // [64, 64]

#define EW(n)  <<<cdiv((n), 256), 256, 0, stream>>>
#define EDGE(E) <<<cdiv((long)(E) * 32, 256), 256, 0, stream>>>

    // ---- degree / symmetric norm (cites only) ----
    k_zero EW(NP_)(dis, NP_);
    k_deg EW(EC)(cites_dst, dis, EC);
    k_rsqrt1 EW(NP_)(dis, NP_);

    // ---- bf16 conversions (inputs + transposed weights) ----
    k_f2b EW(NP_ * 128)(x_paper, xpb, NP_ * 128);
    k_f2b EW(NL_ * 64)(x_label, xlb, NL_ * 64);
    k_wt EW(128 * 128)(W1g,  w1g_t,  128, 128);
    k_wt EW(128 * 128)(W1ri, w1ri_t, 128, 128);
    k_wt EW(64 * 128)(W1oi, w1oi_t, 64, 128);
    k_wt EW(64 * 128)(W1rr, w1rr_t, 64, 128);
    k_wt EW(128 * 128)(W1or, w1or_t, 128, 128);
    k_wt EW(128 * 64)(W2g,  w2g_t,  128, 64);
    k_wt EW(128 * 64)(W2ri, w2ri_t, 128, 64);
    k_wt EW(128 * 64)(W2oi, w2oi_t, 128, 64);
    k_wt EW(128 * 64)(W2rr, w2rr_t, 128, 64);
    k_wt EW(128 * 64)(W2or, w2or_t, 128, 64);

    // ================= layer 1: papers =================
    // GCNConv over cites: y = xp @ W1g
    gemm(xpb, w1g_t, yg, nullptr, NP_, 128, 128, 0, stream);
    k_zero EW(NP_ * 128)(aggP, NP_ * 128);
    k_scatter128 EDGE(EC)(yg, cites_src, cites_dst, dis, aggP, EC);
    k_gcn_combine EW(NP_ * 128)(aggP, yg, dis, b1g, NP_, 128);   // aggP = p_cites
    // GraphConv over rev (labels -> papers)
    k_zero EW(NP_ * 64)(aggRv, NP_ * 64);
    k_scatter64 EDGE(ER)(x_label, rev_src, rev_dst, nullptr, aggRv, ER);
    k_f2b EW(NP_ * 64)(aggRv, aggRvB, NP_ * 64);
    gemm(xpb, w1or_t, prev, b1r, NP_, 128, 128, 0, stream);
    gemm(aggRvB, w1rr_t, prev, nullptr, NP_, 128, 64, 1, stream);
    // mean over relations + ReLU -> xp1 (f32 + bf16)
    k_mean_relu EW(NP_ * 128)(aggP, prev, xp1, xp1b, NP_ * 128);

    // ================= layer 1: labels =================
    k_zero EW(NL_ * 128)(aggIs, NL_ * 128);
    k_scatter128 EDGE(EI)(x_paper, is_src, is_dst, nullptr, aggIs, EI);
    k_f2b EW(NL_ * 128)(aggIs, aggIsB, NL_ * 128);
    gemm(aggIsB, w1ri_t, xl1, b1i, NL_, 128, 128, 0, stream);
    gemm(xlb, w1oi_t, xl1, nullptr, NL_, 128, 64, 1, stream);
    k_relu_conv EW(NL_ * 128)(xl1, xl1b, NL_ * 128);

    // ================= layer 2: papers =================
    gemm(xp1b, w2g_t, yg, nullptr, NP_, 64, 128, 0, stream);
    k_zero EW(NP_ * 64)(aggP, NP_ * 64);
    k_scatter64 EDGE(EC)(yg, cites_src, cites_dst, dis, aggP, EC);
    k_gcn_combine EW(NP_ * 64)(aggP, yg, dis, b2g, NP_, 64);     // aggP = p_cites2
    k_zero EW(NP_ * 128)(aggRv, NP_ * 128);
    k_scatter128 EDGE(ER)(xl1, rev_src, rev_dst, nullptr, aggRv, ER);
    k_f2b EW(NP_ * 128)(aggRv, aggRvB, NP_ * 128);
    gemm(xp1b, w2or_t, prev, b2r, NP_, 64, 128, 0, stream);
    gemm(aggRvB, w2rr_t, prev, nullptr, NP_, 64, 128, 1, stream);
    k_mean EW(NP_ * 64)(aggP, prev, out_p, NP_ * 64);            // final paper output

    // ================= layer 2: labels =================
    k_zero EW(NL_ * 128)(aggIs, NL_ * 128);
    k_scatter128 EDGE(EI)(xp1, is_src, is_dst, nullptr, aggIs, EI);
    k_f2b EW(NL_ * 128)(aggIs, aggIsB, NL_ * 128);
    gemm(aggIsB, w2ri_t, out_l, b2i, NL_, 64, 128, 0, stream);
    gemm(xl1b, w2oi_t, out_l, nullptr, NL_, 64, 128, 1, stream); // final label output

#undef EW
#undef EDGE
}
